// CausalTimeSeriesTransformer_1709396984228
// MI455X (gfx1250) — matrix operand order repaired and checked by reading, not compiled, not run on hardware
//
#include <hip/hip_runtime.h>
#include <hip/hip_bf16.h>
#include <math.h>

#define D_MODEL 512
#define NHEAD   8
#define NLAYER  6
#define FFDIM   2048
#define INDIM   32
#define OUTDIM  3
#define SEQ     512
#define BATCH   32
#define NTOK    (BATCH*SEQ)   /* 16384 */
#define HD      64            /* head dim */

typedef _Float16 half8  __attribute__((ext_vector_type(8)));
typedef _Float16 half16 __attribute__((ext_vector_type(16)));
typedef float    float8 __attribute__((ext_vector_type(8)));

static __device__ __forceinline__ float gelu_f(float x){
  return 0.5f * x * (1.0f + erff(x * 0.7071067811865476f));
}

// Sinusoidal positional encoding value for (position s, channel d)
static __device__ __forceinline__ float pe_val(int s, int d){
  int i = d >> 1;
  float div = expf(-(float)(2 * i) * (9.210340371976184f / (float)D_MODEL)); // ln(10000)=9.2103...
  float a = (float)s * div;
  return (d & 1) ? cosf(a) : sinf(a);
}

// --- WMMA fragment loaders (wave32, 16x16x32 f16) -------------------------
// A-matrix 16x32: lanes 0-15 row M=lane, elems 0..15 = K 0..15
//                 lanes 16-31 row M=lane-16, elems 0..15 = K 16..31
// rowp points at this lane's row base (row stride handled by caller).
static __device__ __forceinline__ half16 load_a32(const _Float16* rowp, int k0, int lane){
  return *(const half16*)(rowp + k0 + ((lane >> 4) << 4));
}
// B-matrix 32x16: lane holds column N=lane&15; lanes 0-15 K=0..15, lanes 16-31 K=16..31.
// rowp points at the Bt (N-major, K contiguous) row for column (lane&15).
static __device__ __forceinline__ half16 load_b32(const _Float16* rowp, int k0, int lane){
  return *(const half16*)(rowp + k0 + ((lane >> 4) << 4));
}
static __device__ __forceinline__ float8 wmma_f16(half16 a, half16 b, float8 c){
  return __builtin_amdgcn_wmma_f32_16x16x32_f16(false, a, false, b, (short)0, c, false, false);
}

// --- fp32 -> f16 transposed weight conversion: Wt[n*K+k] = W[k*N+n] --------
__global__ void convt_kernel(const float* __restrict__ W, _Float16* __restrict__ Wt,
                             int K, int N){
  size_t base  = (size_t)blockIdx.z * (size_t)K * (size_t)N;
  size_t total = (size_t)K * (size_t)N;
  for (size_t idx = (size_t)blockIdx.x * blockDim.x + threadIdx.x; idx < total;
       idx += (size_t)gridDim.x * blockDim.x){
    size_t k = idx / (size_t)N, n = idx % (size_t)N;
    Wt[base + n * (size_t)K + k] = (_Float16)W[base + idx];
  }
}

// --- Embedding: h = gelu(LN(x@w_in + b_in)) + PE ; also writes f16 copy ----
__global__ __launch_bounds__(256) void embed_kernel(
    const float* __restrict__ x, const float* __restrict__ w_in,
    const float* __restrict__ b_in, const float* __restrict__ g_in,
    const float* __restrict__ be_in, float* __restrict__ h,
    _Float16* __restrict__ h16){
  int row = blockIdx.x;                 // token index = b*SEQ + s
  int s   = row & (SEQ - 1);
  int tid = threadIdx.x;                // 256 threads, 2 channels each
  __shared__ float xv[INDIM];
  __shared__ float rsum[256];
  if (tid < INDIM) xv[tid] = x[(size_t)row * INDIM + tid];
  __syncthreads();
  float y0 = b_in[tid], y1 = b_in[tid + 256];
#pragma unroll 8
  for (int k = 0; k < INDIM; k++){
    float xk = xv[k];
    y0 += xk * w_in[k * D_MODEL + tid];
    y1 += xk * w_in[k * D_MODEL + tid + 256];
  }
  rsum[tid] = y0 + y1;
  __syncthreads();
  for (int st = 128; st > 0; st >>= 1){ if (tid < st) rsum[tid] += rsum[tid + st]; __syncthreads(); }
  float mu = rsum[0] * (1.0f / D_MODEL);
  __syncthreads();
  float d0 = y0 - mu, d1 = y1 - mu;
  rsum[tid] = d0*d0 + d1*d1;
  __syncthreads();
  for (int st = 128; st > 0; st >>= 1){ if (tid < st) rsum[tid] += rsum[tid + st]; __syncthreads(); }
  float rstd = rsqrtf(rsum[0] * (1.0f / D_MODEL) + 1e-5f);
  float o0 = gelu_f(d0 * rstd * g_in[tid]       + be_in[tid])       + pe_val(s, tid);
  float o1 = gelu_f(d1 * rstd * g_in[tid + 256] + be_in[tid + 256]) + pe_val(s, tid + 256);
  size_t base = (size_t)row * D_MODEL;
  h[base + tid] = o0;                h[base + tid + 256] = o1;
  h16[base + tid] = (_Float16)o0;    h16[base + tid + 256] = (_Float16)o1;
}

// --- Residual + LayerNorm, writes fp32 state and f16 copy ------------------
__global__ __launch_bounds__(256) void resln_kernel(
    float* __restrict__ h, const float* __restrict__ add,
    const float* __restrict__ g, const float* __restrict__ be,
    _Float16* __restrict__ h16){
  int row = blockIdx.x; int tid = threadIdx.x;
  size_t base = (size_t)row * D_MODEL;
  float v0 = h[base + tid]       + add[base + tid];
  float v1 = h[base + tid + 256] + add[base + tid + 256];
  __shared__ float rsum[256];
  rsum[tid] = v0 + v1;
  __syncthreads();
  for (int st = 128; st > 0; st >>= 1){ if (tid < st) rsum[tid] += rsum[tid + st]; __syncthreads(); }
  float mu = rsum[0] * (1.0f / D_MODEL);
  __syncthreads();
  float d0 = v0 - mu, d1 = v1 - mu;
  rsum[tid] = d0*d0 + d1*d1;
  __syncthreads();
  for (int st = 128; st > 0; st >>= 1){ if (tid < st) rsum[tid] += rsum[tid + st]; __syncthreads(); }
  float rstd = rsqrtf(rsum[0] * (1.0f / D_MODEL) + 1e-5f);
  float y0 = d0 * rstd * g[tid]       + be[tid];
  float y1 = d1 * rstd * g[tid + 256] + be[tid + 256];
  h[base + tid] = y0;                h[base + tid + 256] = y1;
  h16[base + tid] = (_Float16)y0;    h16[base + tid + 256] = (_Float16)y1;
}

// --- Generic WMMA GEMM: block = 256 thr = 8 waves, tile 64 rows x 64 cols --
// wave (rw = wid&3) -> 16 rows ; (cw = wid>>2) -> two 16-col tiles (32 cols)
__global__ __launch_bounds__(256) void gemm_f32_kernel(
    const _Float16* __restrict__ A, const _Float16* __restrict__ Bt,
    const float* __restrict__ bias, float* __restrict__ C, int K, int N){
  int tid = threadIdx.x, lane = tid & 31, wid = tid >> 5;
  int row0 = blockIdx.x * 64 + (wid & 3) * 16;
  int n0   = blockIdx.y * 64 + (wid >> 2) * 32;
  const _Float16* Ap  = A  + (size_t)(row0 + (lane & 15)) * K;
  const _Float16* Bp0 = Bt + (size_t)(n0   + (lane & 15)) * K;
  const _Float16* Bp1 = Bp0 + (size_t)16 * K;
  float8 cacc[2] = {};
  for (int k0 = 0; k0 < K; k0 += 32){
    half16 a  = load_a32(Ap,  k0, lane);
    half16 b0 = load_b32(Bp0, k0, lane);
    half16 b1 = load_b32(Bp1, k0, lane);
    cacc[0] = wmma_f16(a, b0, cacc[0]);
    cacc[1] = wmma_f16(a, b1, cacc[1]);
  }
#pragma unroll
  for (int t = 0; t < 2; t++){
    int col = n0 + t * 16 + (lane & 15);
    float bv = bias[col];
#pragma unroll
    for (int v = 0; v < 8; v++){
      int r = row0 + v + ((lane >> 4) << 3);
      C[(size_t)r * N + col] = cacc[t][v] + bv;
    }
  }
}

__global__ __launch_bounds__(256) void gemm_gelu16_kernel(
    const _Float16* __restrict__ A, const _Float16* __restrict__ Bt,
    const float* __restrict__ bias, _Float16* __restrict__ C, int K, int N){
  int tid = threadIdx.x, lane = tid & 31, wid = tid >> 5;
  int row0 = blockIdx.x * 64 + (wid & 3) * 16;
  int n0   = blockIdx.y * 64 + (wid >> 2) * 32;
  const _Float16* Ap  = A  + (size_t)(row0 + (lane & 15)) * K;
  const _Float16* Bp0 = Bt + (size_t)(n0   + (lane & 15)) * K;
  const _Float16* Bp1 = Bp0 + (size_t)16 * K;
  float8 cacc[2] = {};
  for (int k0 = 0; k0 < K; k0 += 32){
    half16 a  = load_a32(Ap,  k0, lane);
    half16 b0 = load_b32(Bp0, k0, lane);
    half16 b1 = load_b32(Bp1, k0, lane);
    cacc[0] = wmma_f16(a, b0, cacc[0]);
    cacc[1] = wmma_f16(a, b1, cacc[1]);
  }
#pragma unroll
  for (int t = 0; t < 2; t++){
    int col = n0 + t * 16 + (lane & 15);
    float bv = bias[col];
#pragma unroll
    for (int v = 0; v < 8; v++){
      int r = row0 + v + ((lane >> 4) << 3);
      C[(size_t)r * N + col] = (_Float16)gelu_f(cacc[t][v] + bv);
    }
  }
}

// QKV GEMM (N = 3*D fixed): scatter epilogue -> q,k as (b,h,s,d) f16; v as (b,h,d,s) f16
__global__ __launch_bounds__(256) void gemm_qkv_kernel(
    const _Float16* __restrict__ A, const _Float16* __restrict__ Bt,
    const float* __restrict__ bias, _Float16* __restrict__ qh,
    _Float16* __restrict__ kh, _Float16* __restrict__ vT, int K){
  int tid = threadIdx.x, lane = tid & 31, wid = tid >> 5;
  int row0 = blockIdx.x * 64 + (wid & 3) * 16;
  int n0   = blockIdx.y * 64 + (wid >> 2) * 32;
  const _Float16* Ap  = A  + (size_t)(row0 + (lane & 15)) * K;
  const _Float16* Bp0 = Bt + (size_t)(n0   + (lane & 15)) * K;
  const _Float16* Bp1 = Bp0 + (size_t)16 * K;
  float8 cacc[2] = {};
  for (int k0 = 0; k0 < K; k0 += 32){
    half16 a  = load_a32(Ap,  k0, lane);
    half16 b0 = load_b32(Bp0, k0, lane);
    half16 b1 = load_b32(Bp1, k0, lane);
    cacc[0] = wmma_f16(a, b0, cacc[0]);
    cacc[1] = wmma_f16(a, b1, cacc[1]);
  }
#pragma unroll
  for (int t = 0; t < 2; t++){
    int c = n0 + t * 16 + (lane & 15);          // 0..1535
    float bv = bias[c];
    int which = c >> 9;                          // 0:q 1:k 2:v
    int hh    = (c >> 6) & (NHEAD - 1);
    int dd    = c & (HD - 1);
#pragma unroll
    for (int v = 0; v < 8; v++){
      int r = row0 + v + ((lane >> 4) << 3);     // token row = b*SEQ + s
      int b = r >> 9;
      int s = r & (SEQ - 1);
      _Float16 hv = (_Float16)(cacc[t][v] + bv);
      size_t bh = (size_t)(b * NHEAD + hh);
      if (which == 0)      qh[(bh * SEQ + s) * HD + dd] = hv;
      else if (which == 1) kh[(bh * SEQ + s) * HD + dd] = hv;
      else                 vT[(bh * HD + dd) * SEQ + s] = hv;
    }
  }
}

// --- Attention: one block per (b, h, 16-row i-block). 128 thr = 4 waves ----
// Phase 1: scores (16 x valid cols) via WMMA -> LDS (scaled + causal masked)
// Phase 2: row softmax in LDS, emit f16 P (zero-padded to 32-col chunk)
// Phase 3: O = P x V via WMMA (A-fragments from LDS), write (b,s, h*64+d) f16
__global__ __launch_bounds__(128) void attn_kernel(
    const _Float16* __restrict__ qh, const _Float16* __restrict__ kh,
    const _Float16* __restrict__ vT, _Float16* __restrict__ o16){
  __shared__ float    sc[16 * SEQ]  __attribute__((aligned(64)));
  __shared__ _Float16 p16[16 * SEQ] __attribute__((aligned(64)));
  __shared__ float redm[16 * 8];
  __shared__ float reds[16 * 8];

  int blk = blockIdx.x;
  int ib  = blk & (SEQ / 16 - 1);       // i-block 0..31
  int bh  = blk >> 5;                   // b*NHEAD + h
  int b   = bh >> 3;
  int hh  = bh & (NHEAD - 1);
  int i0  = ib * 16;
  int tid = threadIdx.x, lane = tid & 31, wid = tid >> 5;
  int cmax = (((i0 + 16) + 31) >> 5) << 5;   // valid cols rounded up to 32

  const _Float16* qbase = qh + (size_t)bh * SEQ * HD;
  const _Float16* kbase = kh + (size_t)bh * SEQ * HD;

  // Phase 1 ---------------------------------------------------------------
  const _Float16* Qrow = qbase + (size_t)(i0 + (lane & 15)) * HD;
  half16 qa0 = load_a32(Qrow, 0,  lane);
  half16 qa1 = load_a32(Qrow, 32, lane);
  for (int jt = wid; jt <= ib; jt += 4){
    int j0 = jt * 16;
    const _Float16* Krow = kbase + (size_t)(j0 + (lane & 15)) * HD;
    half16 kb0 = load_b32(Krow, 0,  lane);
    half16 kb1 = load_b32(Krow, 32, lane);
    float8 c = {};
    c = wmma_f16(qa0, kb0, c);
    c = wmma_f16(qa1, kb1, c);
    int col = j0 + (lane & 15);
#pragma unroll
    for (int v = 0; v < 8; v++){
      int r = v + ((lane >> 4) << 3);
      float val = c[v] * 0.125f;                 // 1/sqrt(HD)
      if (col > i0 + r) val = -__builtin_inff(); // causal mask
      sc[r * SEQ + col] = val;
    }
  }
  __syncthreads();

  // Phase 2 ---------------------------------------------------------------
  {
    int rowi = tid >> 3, sub = tid & 7;
    int len = i0 + rowi + 1;                     // valid length of this row
    float* srow = &sc[rowi * SEQ];
    float m = -3.0e38f;
    for (int c = sub; c < len; c += 8) m = fmaxf(m, srow[c]);
    redm[rowi * 8 + sub] = m;
    __syncthreads();
    if (sub == 0){
      float mm = redm[rowi * 8];
#pragma unroll
      for (int i = 1; i < 8; i++) mm = fmaxf(mm, redm[rowi * 8 + i]);
      redm[rowi * 8] = mm;
    }
    __syncthreads();
    float mx = redm[rowi * 8];
    float ss = 0.0f;
    for (int c = sub; c < len; c += 8){
      float e = expf(srow[c] - mx);
      srow[c] = e;
      ss += e;
    }
    reds[rowi * 8 + sub] = ss;
    __syncthreads();
    if (sub == 0){
      float t = 0.0f;
#pragma unroll
      for (int i = 0; i < 8; i++) t += reds[rowi * 8 + i];
      reds[rowi * 8] = t;
    }
    __syncthreads();
    float inv = 1.0f / reds[rowi * 8];
    _Float16* prow = &p16[rowi * SEQ];
    for (int c = sub; c < len; c += 8)      prow[c] = (_Float16)(srow[c] * inv);
    for (int c = len + sub; c < cmax; c += 8) prow[c] = (_Float16)0.0f;
  }
  __syncthreads();

  // Phase 3 ---------------------------------------------------------------
  {
    int d0 = wid * 16;                           // each wave: 16 hd columns
    const _Float16* Vrow = vT + ((size_t)bh * HD + d0 + (lane & 15)) * SEQ;
    const _Float16* Prow = &p16[(lane & 15) * SEQ];
    float8 c = {};
    for (int kc = 0; kc < cmax; kc += 32){
      half16 pa = load_a32(Prow, kc, lane);      // A fragment from LDS
      half16 vb = load_b32(Vrow, kc, lane);
      c = wmma_f16(pa, vb, c);
    }
    int ch = hh * HD + d0 + (lane & 15);
#pragma unroll
    for (int v = 0; v < 8; v++){
      int r = i0 + v + ((lane >> 4) << 3);
      o16[((size_t)(b * SEQ + r)) * D_MODEL + ch] = (_Float16)c[v];
    }
  }
}

// --- Head: last token per batch: out = gelu(h@w1+b1)@w2+b2 ----------------
__global__ __launch_bounds__(256) void head_kernel(
    const float* __restrict__ h, const float* __restrict__ w1,
    const float* __restrict__ b1, const float* __restrict__ w2,
    const float* __restrict__ b2, float* __restrict__ out){
  int b = blockIdx.x; int tid = threadIdx.x;     // 256 threads = 256 hidden
  __shared__ float gbuf[256];
  const float* hr = h + ((size_t)(b * SEQ + SEQ - 1)) * D_MODEL;
  float acc = b1[tid];
  for (int k = 0; k < D_MODEL; k++) acc += hr[k] * w1[k * 256 + tid];
  gbuf[tid] = gelu_f(acc);
  __syncthreads();
  if (tid < OUTDIM){
    float a = b2[tid];
    for (int k = 0; k < 256; k++) a += gbuf[k] * w2[k * OUTDIM + tid];
    out[b * OUTDIM + tid] = a;
  }
}

extern "C" void kernel_launch(void* const* d_in, const int* in_sizes, int n_in,
                              void* d_out, int out_size, void* d_ws, size_t ws_size,
                              hipStream_t stream) {
  (void)in_sizes; (void)n_in; (void)out_size; (void)ws_size;
  const float* x     = (const float*)d_in[0];
  const float* w_in  = (const float*)d_in[1];
  const float* b_in  = (const float*)d_in[2];
  const float* g_in  = (const float*)d_in[3];
  const float* be_in = (const float*)d_in[4];
  const float* w_qkv = (const float*)d_in[5];
  const float* b_qkv = (const float*)d_in[6];
  const float* w_out = (const float*)d_in[7];
  const float* b_out = (const float*)d_in[8];
  const float* g1    = (const float*)d_in[9];
  const float* be1   = (const float*)d_in[10];
  const float* w_ff1 = (const float*)d_in[11];
  const float* b_ff1 = (const float*)d_in[12];
  const float* w_ff2 = (const float*)d_in[13];
  const float* b_ff2 = (const float*)d_in[14];
  const float* g2    = (const float*)d_in[15];
  const float* be2   = (const float*)d_in[16];
  const float* w_h1  = (const float*)d_in[17];
  const float* b_h1  = (const float*)d_in[18];
  const float* w_h2  = (const float*)d_in[19];
  const float* b_h2  = (const float*)d_in[20];

  // ---- workspace carve (all 256B aligned) ----
  char* ws = (char*)d_ws;
  size_t off = 0;
  auto carve = [&](size_t bytes) -> char* {
    char* p = ws + off;
    off += (bytes + 255) & ~(size_t)255;
    return p;
  };
  _Float16* wt_qkv = (_Float16*)carve((size_t)NLAYER * 3 * D_MODEL * D_MODEL * sizeof(_Float16));
  _Float16* wt_out = (_Float16*)carve((size_t)NLAYER * D_MODEL * D_MODEL * sizeof(_Float16));
  _Float16* wt_ff1 = (_Float16*)carve((size_t)NLAYER * D_MODEL * FFDIM * sizeof(_Float16));
  _Float16* wt_ff2 = (_Float16*)carve((size_t)NLAYER * FFDIM * D_MODEL * sizeof(_Float16));
  float*    h      = (float*)   carve((size_t)NTOK * D_MODEL * sizeof(float));
  _Float16* h16    = (_Float16*)carve((size_t)NTOK * D_MODEL * sizeof(_Float16));
  float*    t32    = (float*)   carve((size_t)NTOK * D_MODEL * sizeof(float));
  _Float16* R1     = (_Float16*)carve((size_t)NTOK * FFDIM * sizeof(_Float16));
  // alias layout inside R1 (lifetimes disjoint with t16):
  _Float16* qhb = R1;
  _Float16* khb = R1 + (size_t)NTOK * D_MODEL;
  _Float16* vTb = R1 + (size_t)2 * NTOK * D_MODEL;
  _Float16* o16 = R1 + (size_t)3 * NTOK * D_MODEL;
  _Float16* t16 = R1;

  // ---- weights: fp32 -> f16, transposed to (N, K) ----
  convt_kernel<<<dim3(512, 1, NLAYER), 256, 0, stream>>>(w_qkv, wt_qkv, D_MODEL, 3 * D_MODEL);
  convt_kernel<<<dim3(512, 1, NLAYER), 256, 0, stream>>>(w_out, wt_out, D_MODEL, D_MODEL);
  convt_kernel<<<dim3(512, 1, NLAYER), 256, 0, stream>>>(w_ff1, wt_ff1, D_MODEL, FFDIM);
  convt_kernel<<<dim3(512, 1, NLAYER), 256, 0, stream>>>(w_ff2, wt_ff2, FFDIM, D_MODEL);

  // ---- embedding ----
  embed_kernel<<<NTOK, 256, 0, stream>>>(x, w_in, b_in, g_in, be_in, h, h16);

  // ---- transformer layers ----
  for (int l = 0; l < NLAYER; l++){
    gemm_qkv_kernel<<<dim3(NTOK / 64, (3 * D_MODEL) / 64), 256, 0, stream>>>(
        h16, wt_qkv + (size_t)l * 3 * D_MODEL * D_MODEL,
        b_qkv + (size_t)l * 3 * D_MODEL, qhb, khb, vTb, D_MODEL);

    attn_kernel<<<BATCH * NHEAD * (SEQ / 16), 128, 0, stream>>>(qhb, khb, vTb, o16);

    gemm_f32_kernel<<<dim3(NTOK / 64, D_MODEL / 64), 256, 0, stream>>>(
        o16, wt_out + (size_t)l * D_MODEL * D_MODEL,
        b_out + (size_t)l * D_MODEL, t32, D_MODEL, D_MODEL);

    resln_kernel<<<NTOK, 256, 0, stream>>>(h, t32, g1 + l * D_MODEL, be1 + l * D_MODEL, h16);

    gemm_gelu16_kernel<<<dim3(NTOK / 64, FFDIM / 64), 256, 0, stream>>>(
        h16, wt_ff1 + (size_t)l * D_MODEL * FFDIM,
        b_ff1 + (size_t)l * FFDIM, t16, D_MODEL, FFDIM);

    gemm_f32_kernel<<<dim3(NTOK / 64, D_MODEL / 64), 256, 0, stream>>>(
        t16, wt_ff2 + (size_t)l * FFDIM * D_MODEL,
        b_ff2 + (size_t)l * D_MODEL, t32, FFDIM, D_MODEL);

    resln_kernel<<<NTOK, 256, 0, stream>>>(h, t32, g2 + l * D_MODEL, be2 + l * D_MODEL, h16);
  }

  // ---- head ----
  head_kernel<<<BATCH, 256, 0, stream>>>(h, w_h1, b_h1, w_h2, b_h2, (float*)d_out);
}
